// MegatronBaselineMoE_67903432950548
// MI455X (gfx1250) — matrix operand order repaired and checked by reading, not compiled, not run on hardware
//
#include <hip/hip_runtime.h>
#include <hip/hip_bf16.h>
#include <math.h>
#include <stdint.h>

#define N_TOK 2048
#define H_DIM 1024
#define F_DIM 2048
#define N_EXP 8

#define KP1 1032   // w1t row stride (1024 k + 8 pad) in shorts
#define KP2 2056   // w2t row stride (2048 k + 8 pad) in shorts
#define XP_ROWS 4224  // 4096 routed slots + 128 pad rows

// ---- workspace layout (bytes) ----
#define OFF_PROBS  0ull
#define OFF_COUNTS 65536ull
#define OFF_PREFIX 65600ull
#define OFF_GATH   65792ull
#define OFF_XP     262144ull
#define OFF_W1T    (OFF_XP + (unsigned long long)XP_ROWS * H_DIM * 2)        // + 8.65MB
#define OFF_W2T    (OFF_W1T + 8ull * F_DIM * KP1 * 2)                        // + 33.8MB
#define OFF_H      (OFF_W2T + 8ull * H_DIM * KP2 * 2)                        // + 33.7MB

typedef __attribute__((ext_vector_type(16))) __bf16 v16bf;
typedef __attribute__((ext_vector_type(2)))  __bf16 v2bf;
typedef __attribute__((ext_vector_type(8)))  float  v8f;
typedef __attribute__((ext_vector_type(4)))  unsigned int v4u;
typedef __attribute__((ext_vector_type(8)))  int v8i;
typedef __attribute__((ext_vector_type(4)))  int v4i;

#if __has_builtin(__builtin_amdgcn_tensor_load_to_lds) && __has_builtin(__builtin_amdgcn_s_wait_tensorcnt)
#define USE_TDM 1
#else
#define USE_TDM 0
#endif

// ---- f32 -> bf16 (RNE) --------------------------------------------------
__device__ __forceinline__ unsigned short f2bf(float f) {
  unsigned int u = __float_as_uint(f);
  u += 0x7FFFu + ((u >> 16) & 1u);
  return (unsigned short)(u >> 16);
}
__device__ __forceinline__ unsigned int pkbf(float a, float b) {
#if __has_builtin(__builtin_amdgcn_cvt_pk_bf16_f32)
  union { v2bf v; unsigned int u; } c;
  c.v = __builtin_amdgcn_cvt_pk_bf16_f32(a, b);
  return c.u;
#else
  return (unsigned int)f2bf(a) | ((unsigned int)f2bf(b) << 16);
#endif
}

// ---- WMMA fragment loads (wave32 layouts, CDNA5 ISA) ---------
// A 16x32: lane L (row=L&15, half=L>>4): elems 0..7 = K[k0+8h..+7],
// elems 8..15 = K[k0+16+8h..+7]; p = rowbase + k0 + 8*half.
__device__ __forceinline__ v16bf load_a_frag(const unsigned short* p) {
  union { v16bf v; uint4 q[2]; } u;
  u.q[0] = *(const uint4*)(p);
  u.q[1] = *(const uint4*)(p + 16);
  return u.v;
}
// B 32x16: lane L (col=L&15, half=L>>4): elems 0..7 = K[k0+8h..+7],
// elems 8..15 = K[k0+16+8h..+7] of column col (works for LDS or global pointers).
__device__ __forceinline__ v16bf load_b_frag(const unsigned short* p) {
  union { v16bf v; uint4 q[2]; } u;
  u.q[0] = *(const uint4*)(p);
  u.q[1] = *(const uint4*)(p + 16);
  return u.v;
}

// ---- TDM: 128-row x 256-byte 2D tile -> LDS (row pad 256B->272B) --------
__device__ __forceinline__ void tdm_issue(void* lds_dst, const void* gsrc,
                                          unsigned row_stride_bytes) {
#if USE_TDM
  unsigned long long ga = (unsigned long long)(uintptr_t)gsrc;
  unsigned lds_off = (unsigned)(uintptr_t)lds_dst;   // low 32 bits = LDS offset
  v4u g0 = {1u,                                   // count=1, user descriptor
            lds_off,
            (unsigned)ga,
            (unsigned)((ga >> 32) & 0x1FFFFFFu) | (2u << 30)};  // type=2
  const unsigned td0 = 0x40000000u, td1 = 0x40000000u;  // huge dims: no OOB clip
  v8i g1 = {(int)((1u << 20) | (5u << 22) | (3u << 25)), // 1B elems, pad 16B/256B
            (int)((td0 & 0xFFFFu) << 16),                // [47:32]=0, dim0 lo
            (int)((td0 >> 16) | ((td1 & 0xFFFFu) << 16)),
            (int)((td1 >> 16) | (256u << 16)),           // tile_dim0 = 256 B
            (int)128,                                    // tile_dim1 = 128 rows
            (int)row_stride_bytes,                       // dim0 stride (48b lo)
            0, 0};
  v4i g2 = {0, 0, 0, 0};
  v4i g3 = {0, 0, 0, 0};
#if defined(__clang_major__) && (__clang_major__ >= 23)
  v8i g4 = {0, 0, 0, 0, 0, 0, 0, 0};
  __builtin_amdgcn_tensor_load_to_lds(g0, g1, g2, g3, g4, 0);
#else
  __builtin_amdgcn_tensor_load_to_lds(g0, g1, g2, g3, 0);
#endif
#endif
}

// Sync fallback: cooperative copy of the same tile (512 threads).
__device__ __forceinline__ void copy_tile(unsigned short* dst, const unsigned short* src,
                                          int src_stride_shorts, int tid) {
  const int row = tid >> 2, q = tid & 3;              // 4 threads x 32 shorts per row
  const uint4* g = (const uint4*)(src + (size_t)row * src_stride_shorts + q * 32);
  uint4* l = (uint4*)(dst + row * 136 + q * 32);
#pragma unroll
  for (int i = 0; i < 4; ++i) l[i] = g[i];
}

// ---- GEMM inner block with rotating B-fragment prefetch -----------------
// Computes 8 m-tiles x 1 n-tile x 128 K; `b` holds the current B fragment
// and is replaced by the next k-step's fragment (prefetched before the
// 8-WMMA group so its L2 latency hides under the matrix ops). The one-past-
// the-end prefetch over-reads <=48B into the adjacent ws region (harmless).
__device__ __forceinline__ void gemm_block_pf(const unsigned short* Ab,  // LDS [128][136]
                                              const unsigned short* bb,  // global col base
                                              int kb, int col16, int half,
                                              v8f acc[8], v16bf& b) {
#pragma unroll
  for (int ks = 0; ks < 4; ++ks) {
    v16bf bn = load_b_frag(bb + kb * 128 + ks * 32 + 32 + 8 * half);
#pragma unroll
    for (int mt = 0; mt < 8; ++mt) {
      v16bf a = load_a_frag(Ab + (mt * 16 + col16) * 136 + ks * 32 + 8 * half);
      acc[mt] = __builtin_amdgcn_wmma_f32_16x16x32_bf16(
          false, a, false, b, (short)0, acc[mt], false, false);
    }
    b = bn;
  }
}

// ---------------- Router -------------------------------------------------
__global__ void moe_router(const float* __restrict__ x, const float* __restrict__ rw,
                           float* __restrict__ probs, int* __restrict__ counts,
                           int* __restrict__ gathered) {
  const int wave = threadIdx.x >> 5, lane = threadIdx.x & 31;
  const int t = blockIdx.x * 8 + wave;
  float acc[N_EXP];
#pragma unroll
  for (int e = 0; e < N_EXP; ++e) acc[e] = 0.f;
  const float* xr = x + (size_t)t * H_DIM;
  for (int h = lane; h < H_DIM; h += 32) {
    float xv = xr[h];
    const float* r = rw + h * N_EXP;
#pragma unroll
    for (int e = 0; e < N_EXP; ++e) acc[e] += xv * r[e];
  }
#pragma unroll
  for (int off = 16; off >= 1; off >>= 1)
#pragma unroll
    for (int e = 0; e < N_EXP; ++e) acc[e] += __shfl_xor(acc[e], off, 32);
  if (lane == 0) {
    float m = acc[0];
#pragma unroll
    for (int e = 1; e < N_EXP; ++e) m = fmaxf(m, acc[e]);
    float p[N_EXP], s = 0.f;
#pragma unroll
    for (int e = 0; e < N_EXP; ++e) { p[e] = __expf(acc[e] - m); s += p[e]; }
    float inv = 1.f / s;
#pragma unroll
    for (int e = 0; e < N_EXP; ++e) p[e] *= inv;
    int i1 = 0; float p1 = p[0];
#pragma unroll
    for (int e = 1; e < N_EXP; ++e) if (p[e] > p1) { p1 = p[e]; i1 = e; }
    int i2 = (i1 == 0) ? 1 : 0; float p2 = p[i2];
#pragma unroll
    for (int e = 0; e < N_EXP; ++e) if (e != i1 && p[e] > p2) { p2 = p[e]; i2 = e; }
    float rs = 1.f / (p1 + p2);
#pragma unroll
    for (int e = 0; e < N_EXP; ++e)
      probs[t * N_EXP + e] = (e == i1) ? p1 * rs : ((e == i2) ? p2 * rs : 0.f);
    int pos = atomicAdd(&counts[i1], 1); gathered[i1 * N_TOK + pos] = t;
    pos     = atomicAdd(&counts[i2], 1); gathered[i2 * N_TOK + pos] = t;
  }
}

__global__ void moe_prefix(const int* __restrict__ counts, int* __restrict__ prefix) {
  if (threadIdx.x == 0) {
    int s = 0;
#pragma unroll
    for (int e = 0; e < N_EXP; ++e) { prefix[e] = s; s += counts[e]; }
  }
}

// ---- gather + convert x rows into compact per-expert bf16 matrix --------
__global__ void __launch_bounds__(256)
moe_gather_x(const float* __restrict__ x, const int* __restrict__ counts,
             const int* __restrict__ prefix, const int* __restrict__ gathered,
             unsigned short* __restrict__ xp) {
  const int e = blockIdx.x >> 6, it = blockIdx.x & 63;
  const int cnt = counts[e];
  const int row = it * 32 + (threadIdx.x >> 3);
  if (row >= cnt) return;
  const int seg = threadIdx.x & 7;
  const int token = gathered[e * N_TOK + row];
  const float4* s = (const float4*)(x + (size_t)token * H_DIM + seg * 128);
  unsigned short* d = xp + (size_t)(prefix[e] + row) * H_DIM + seg * 128;
#pragma unroll
  for (int i = 0; i < 32; ++i) {
    float4 v = s[i];
    *(uint2*)(d + i * 4) = make_uint2(pkbf(v.x, v.y), pkbf(v.z, v.w));
  }
}

// ---- convert+transpose weights: [K][Nn] f32 -> [Nn][KP] bf16 ------------
__global__ void __launch_bounds__(256)
moe_cvt_t(const float* __restrict__ src, unsigned short* __restrict__ dst,
          int K, int Nn, int KP) {
  const int tilesN = Nn >> 6, tilesK = K >> 6;
  int b = blockIdx.x;
  const int e = b / (tilesN * tilesK); b %= tilesN * tilesK;
  const int kt = b / tilesN, nt = b % tilesN;
  const float* se = src + (size_t)e * K * Nn + (size_t)(kt * 64) * Nn + nt * 64;
  unsigned short* de = dst + (size_t)e * Nn * KP + (size_t)(nt * 64) * KP + kt * 64;
  __shared__ float t[64][65];
  const int r = threadIdx.x >> 2, q = threadIdx.x & 3;
  const float4* s4 = (const float4*)(se + (size_t)r * Nn + q * 16);
#pragma unroll
  for (int i = 0; i < 4; ++i) {
    float4 v = s4[i];
    t[r][q * 16 + i * 4 + 0] = v.x; t[r][q * 16 + i * 4 + 1] = v.y;
    t[r][q * 16 + i * 4 + 2] = v.z; t[r][q * 16 + i * 4 + 3] = v.w;
  }
  __syncthreads();
#pragma unroll
  for (int i = 0; i < 8; ++i) {
    int k = q * 16 + i * 2;
    *(unsigned int*)(de + (size_t)r * KP + k) = pkbf(t[k][r], t[k + 1][r]);
  }
}

// ---- Pass A: h = gelu(xp @ w1t), M=128 x N=256 per block ----------------
__global__ void __launch_bounds__(512)
moe_gemm1(const unsigned short* __restrict__ xp, const unsigned short* __restrict__ w1t,
          const int* __restrict__ counts, const int* __restrict__ prefix,
          unsigned short* __restrict__ h) {
  const int e  = blockIdx.x >> 7;           // [e:8][mt:16][fr:8]
  const int mt = (blockIdx.x >> 3) & 15;
  const int fr = blockIdx.x & 7;
  const int cnt = counts[e], m0 = mt * 128;
  if (m0 >= cnt) return;
  const int rows = (cnt - m0 < 128) ? (cnt - m0) : 128;
  const int base = prefix[e] + m0;
  const int f0 = fr * 256;

  __shared__ __align__(16) unsigned short Abuf[2][128 * 136];

  const int tid = threadIdx.x, wave = tid >> 5, lane = tid & 31;
  const int col16 = lane & 15, half = lane >> 4;
  const unsigned short* asrc = xp + (size_t)base * H_DIM;
  const unsigned short* bb =
      w1t + ((size_t)e * F_DIM + f0 + wave * 16 + col16) * KP1;

  v8f acc[8];
#pragma unroll
  for (int i = 0; i < 8; ++i) acc[i] = (v8f){};
  v16bf bcur = load_b_frag(bb + 8 * half);   // k-step 0 fragment in flight

#if USE_TDM
  if (wave == 0) tdm_issue(&Abuf[0][0], asrc, H_DIM * 2);
#endif
#pragma unroll 1
  for (int kb = 0; kb < 8; ++kb) {
    const int p = kb & 1;
    __syncthreads();  // previous readers of Abuf[p^1] are done
#if USE_TDM
    if (wave == 0) {
      if (kb + 1 < 8) {
        tdm_issue(&Abuf[p ^ 1][0], asrc + (kb + 1) * 128, H_DIM * 2);
        __builtin_amdgcn_s_wait_tensorcnt(1);
      } else {
        __builtin_amdgcn_s_wait_tensorcnt(0);
      }
    }
#else
    copy_tile(&Abuf[p][0], asrc + kb * 128, H_DIM, tid);
#endif
    __syncthreads();  // Abuf[p] ready for all waves
    gemm_block_pf(&Abuf[p][0], bb, kb, col16, half, acc, bcur);
  }

  // erf-GELU, assemble rows in LDS, coalesced bf16 store of h tile.
  unsigned short* hs = &Abuf[0][0];       // reuse as [128][264]
  __syncthreads();
#pragma unroll
  for (int mt2 = 0; mt2 < 8; ++mt2)
#pragma unroll
    for (int r = 0; r < 8; ++r) {
      float v = acc[mt2][r];
      v = 0.5f * v * (1.f + erff(v * 0.70710678118f));
      hs[(mt2 * 16 + r + 8 * half) * 264 + wave * 16 + col16] = f2bf(v);
    }
  __syncthreads();
  {
    const int row = tid >> 2, q = tid & 3;
    if (row < rows) {
      const uint4* s = (const uint4*)(hs + row * 264 + q * 64);
      uint4* d = (uint4*)(h + (size_t)(base + row) * F_DIM + f0 + q * 64);
#pragma unroll
      for (int i = 0; i < 4; ++i) d[i] = s[i];
    }
  }
}

// ---- Pass B: out += gate * (h @ w2t), M=128 x N=256 per block -----------
__global__ void __launch_bounds__(512)
moe_gemm2(const unsigned short* __restrict__ h, const unsigned short* __restrict__ w2t,
          const int* __restrict__ counts, const int* __restrict__ prefix,
          const int* __restrict__ gathered, const float* __restrict__ probs,
          float* __restrict__ out) {
  const int e  = blockIdx.x >> 6;           // [e:8][mt:16][hr:4]
  const int mt = (blockIdx.x >> 2) & 15;
  const int hr = blockIdx.x & 3;
  const int cnt = counts[e], m0 = mt * 128;
  if (m0 >= cnt) return;
  const int rows = (cnt - m0 < 128) ? (cnt - m0) : 128;
  const int base = prefix[e] + m0;
  const int h0 = hr * 256;

  __shared__ __align__(16) unsigned short Abuf[2][128 * 136];
  __shared__ int   tokl[128];
  __shared__ float gatel[128];

  const int tid = threadIdx.x, wave = tid >> 5, lane = tid & 31;
  const int col16 = lane & 15, half = lane >> 4;
  if (tid < 128) {
    int t = (tid < rows) ? gathered[e * N_TOK + m0 + tid] : 0;
    tokl[tid]  = t;
    gatel[tid] = (tid < rows) ? probs[t * N_EXP + e] : 0.f;
  }
  const unsigned short* asrc = h + (size_t)base * F_DIM;
  const unsigned short* bb =
      w2t + ((size_t)e * H_DIM + h0 + wave * 16 + col16) * KP2;

  v8f acc[8];
#pragma unroll
  for (int i = 0; i < 8; ++i) acc[i] = (v8f){};
  v16bf bcur = load_b_frag(bb + 8 * half);

#if USE_TDM
  if (wave == 0) tdm_issue(&Abuf[0][0], asrc, F_DIM * 2);
#endif
#pragma unroll 1
  for (int kb = 0; kb < 16; ++kb) {
    const int p = kb & 1;
    __syncthreads();
#if USE_TDM
    if (wave == 0) {
      if (kb + 1 < 16) {
        tdm_issue(&Abuf[p ^ 1][0], asrc + (kb + 1) * 128, F_DIM * 2);
        __builtin_amdgcn_s_wait_tensorcnt(1);
      } else {
        __builtin_amdgcn_s_wait_tensorcnt(0);
      }
    }
#else
    copy_tile(&Abuf[p][0], asrc + kb * 128, F_DIM, tid);
#endif
    __syncthreads();
    gemm_block_pf(&Abuf[p][0], bb, kb, col16, half, acc, bcur);
  }

  // gate-scale, scatter-add (2 commutative f32 adds per token element)
#pragma unroll
  for (int mt2 = 0; mt2 < 8; ++mt2)
#pragma unroll
    for (int r = 0; r < 8; ++r) {
      const int lm = mt2 * 16 + r + 8 * half;
      if (lm < rows) {
        float g = gatel[lm];
        atomicAdd(&out[(size_t)tokl[lm] * H_DIM + h0 + wave * 16 + col16],
                  g * acc[mt2][r]);
      }
    }
}

extern "C" void kernel_launch(void* const* d_in, const int* in_sizes, int n_in,
                              void* d_out, int out_size, void* d_ws, size_t ws_size,
                              hipStream_t stream) {
  const float* x  = (const float*)d_in[0];
  const float* rw = (const float*)d_in[1];
  const float* w1 = (const float*)d_in[2];
  const float* w2 = (const float*)d_in[3];
  float* out = (float*)d_out;

  char* ws = (char*)d_ws;
  float*          probs    = (float*)(ws + OFF_PROBS);
  int*            counts   = (int*)(ws + OFF_COUNTS);
  int*            prefix   = (int*)(ws + OFF_PREFIX);
  int*            gathered = (int*)(ws + OFF_GATH);
  unsigned short* xp       = (unsigned short*)(ws + OFF_XP);
  unsigned short* w1t      = (unsigned short*)(ws + OFF_W1T);
  unsigned short* w2t      = (unsigned short*)(ws + OFF_W2T);
  unsigned short* hbuf     = (unsigned short*)(ws + OFF_H);

  hipMemsetAsync(counts, 0, N_EXP * sizeof(int), stream);
  hipMemsetAsync(out, 0, (size_t)N_TOK * H_DIM * sizeof(float), stream);

  moe_router<<<N_TOK / 8, 256, 0, stream>>>(x, rw, probs, counts, gathered);
  moe_prefix<<<1, 32, 0, stream>>>(counts, prefix);
  moe_gather_x<<<N_EXP * 64, 256, 0, stream>>>(x, counts, prefix, gathered, xp);
  moe_cvt_t<<<N_EXP * 16 * 32, 256, 0, stream>>>(w1, w1t, H_DIM, F_DIM, KP1);
  moe_cvt_t<<<N_EXP * 32 * 16, 256, 0, stream>>>(w2, w2t, F_DIM, H_DIM, KP2);
  moe_gemm1<<<N_EXP * 16 * 8, 512, 0, stream>>>(xp, w1t, counts, prefix, hbuf);
  moe_gemm2<<<N_EXP * 16 * 4, 512, 0, stream>>>(hbuf, w2t, counts, prefix,
                                                gathered, probs, out);
}
